// Encoder_37022618091968
// MI455X (gfx1250) — matrix operand order repaired and checked by reading, not compiled, not run on hardware
//
#include <hip/hip_runtime.h>
#include <hip/hip_bf16.h>
#include <math.h>

// Problem dims (match reference)
#define B_  2
#define S_  1024
#define D_  512
#define H_  8
#define DK_ 64
#define L_  8
#define FF_ 2048
#define M_  (B_ * S_)          // 2048 activation rows
#define HD_ (H_ * DK_)         // 512

typedef __attribute__((ext_vector_type(16))) __bf16        bf16x16;
typedef __attribute__((ext_vector_type(8)))  float         f32x8;
typedef __attribute__((ext_vector_type(4)))  unsigned int  u32x4;

union FragAB { u32x4 q[2]; bf16x16 v; };   // 32B = 8 VGPRs = one WMMA 16-bit operand

__device__ __forceinline__ unsigned int bf16b(float f) {
  unsigned int u = __float_as_uint(f);
  return (u + 0x7FFFu + ((u >> 16) & 1u)) >> 16;   // RNE f32 -> bf16 bits
}

// gfx1250 async copy: global -> LDS, 16 bytes per lane, tracked by ASYNCcnt.
__device__ __forceinline__ void async_copy_b128(const unsigned short* g,
                                                unsigned short* l) {
  unsigned int lds = (unsigned int)(unsigned long long)l;  // low 32b = LDS offset
  asm volatile("global_load_async_to_lds_b128 %0, %1, off"
               :: "v"(lds), "v"(g) : "memory");
}
__device__ __forceinline__ void wait_async0() {
  asm volatile("s_wait_asynccnt 0x0" ::: "memory");
}
__device__ __forceinline__ void wait_async3() {
  asm volatile("s_wait_asynccnt 0x3" ::: "memory");
}

// Build an A/B fragment from an LDS row (row-major [idx][k], 16B-aligned rows).
__device__ __forceinline__ FragAB frag_a(const unsigned short* row, int half) {
  FragAB f;
  f.q[0] = *(const u32x4*)(row + half * 8);        // K 0..7   (+8 if half)
  f.q[1] = *(const u32x4*)(row + 16 + half * 8);   // K 16..23 (+8 if half)
  return f;
}
__device__ __forceinline__ FragAB frag_b(const unsigned short* row, int half) {
  FragAB f;
  f.q[0] = *(const u32x4*)(row + half * 8);        // K 0..7   (+8 if half)
  f.q[1] = *(const u32x4*)(row + 16 + half * 8);   // K 16..23 (+8 if half)
  return f;
}
#define WMMA_BF16(A, B, C) \
  __builtin_amdgcn_wmma_f32_16x16x32_bf16(false, (A), false, (B), (short)0, (C), false, false)

// ---------------------------------------------------------------------------
// Embedding gather: writes f32 x and bf16 xb
// ---------------------------------------------------------------------------
__global__ void __launch_bounds__(256) embed_kernel(
    const int* __restrict__ tokens, const float* __restrict__ emb,
    float* __restrict__ x, unsigned short* __restrict__ xb)
{
  int idx = blockIdx.x * 256 + threadIdx.x;   // over M_*D_
  int row = idx >> 9;                          // D_ == 512
  int d   = idx & (D_ - 1);
  float v = emb[(size_t)tokens[row] * D_ + d];
  x[idx]  = v;
  xb[idx] = (unsigned short)bf16b(v);
}

// ---------------------------------------------------------------------------
// Weight convert+transpose: in f32 [K][N] -> out bf16 [N][K]  (N = 1<<nlog2)
// ---------------------------------------------------------------------------
__global__ void __launch_bounds__(256) convtrans_kernel(
    const float* __restrict__ in, unsigned short* __restrict__ out,
    int K, int nlog2)
{
  int idx = blockIdx.x * 256 + threadIdx.x;   // over K*N, coalesced over n
  int k = idx >> nlog2;
  int n = idx & ((1 << nlog2) - 1);
  out[(size_t)n * K + k] = (unsigned short)bf16b(in[idx]);
}

// ---------------------------------------------------------------------------
// GEMM: C = act(Ab[M,K](bf16) @ Bt[N,K](bf16)^T + bias (+residual f32))
// 128x64 block tile, BK=32, 8 waves each computing 32x32 (4 WMMAs from
// 8 ds_load_b128: A frags shared across n-subtiles, B frags across m-subtiles).
// LDS double-buffered, filled by global_load_async_to_lds_b128 (3/thread/slab),
// next K-slab streams behind the WMMAs (s_wait_asynccnt 0x3).
// ---------------------------------------------------------------------------
__global__ void __launch_bounds__(256) gemm_bf16_wmma(
    const unsigned short* __restrict__ Ab, const unsigned short* __restrict__ Bt,
    const float* __restrict__ bias, const float* __restrict__ residual,
    float* __restrict__ Cf, unsigned short* __restrict__ Cb,
    unsigned short* __restrict__ CbT,
    int M, int N, int K, int do_relu)
{
  __shared__ __align__(16) unsigned short As[2][128 * 40];  // [m][k], 80B stride
  __shared__ __align__(16) unsigned short Bs[2][64 * 40];   // [n][k]

  const int t = threadIdx.x;
  const int rowBase = blockIdx.y * 128, colBase = blockIdx.x * 64;
  const int lane = t & 31, wid = t >> 5;
  const int half = lane >> 4, l16 = lane & 15;
  const int wm = wid >> 1, wn = wid & 1;      // wave: rows wm*32, cols wn*32
  const int cr = t >> 2, cs = (t & 3) << 3;   // copy row/seg (ushorts)

  auto issue = [&](int kb, int buf) {
    const int kBase = kb << 5;
    async_copy_b128(Ab + (size_t)(rowBase + cr) * K + kBase + cs,
                    &As[buf][cr * 40 + cs]);
    async_copy_b128(Ab + (size_t)(rowBase + cr + 64) * K + kBase + cs,
                    &As[buf][(cr + 64) * 40 + cs]);
    async_copy_b128(Bt + (size_t)(colBase + cr) * K + kBase + cs,
                    &Bs[buf][cr * 40 + cs]);
  };

  f32x8 acc[2][2] = {};   // [msub][nsub]
  const int nkb = K >> 5;
  issue(0, 0);
  for (int kb = 0; kb < nkb; ++kb) {
    const int cur = kb & 1;
    const bool more = (kb + 1 < nkb);
    if (more) issue(kb + 1, cur ^ 1);   // stream next slab behind the math
    if (more) wait_async3(); else wait_async0();
    __syncthreads();

    const FragAB fa0 = frag_a(&As[cur][(wm * 32 +      l16) * 40], half);
    const FragAB fa1 = frag_a(&As[cur][(wm * 32 + 16 + l16) * 40], half);
    const FragAB fb0 = frag_b(&Bs[cur][(wn * 32 +      l16) * 40], half);
    const FragAB fb1 = frag_b(&Bs[cur][(wn * 32 + 16 + l16) * 40], half);

    acc[0][0] = WMMA_BF16(fa0.v, fb0.v, acc[0][0]);
    acc[0][1] = WMMA_BF16(fa0.v, fb1.v, acc[0][1]);
    acc[1][0] = WMMA_BF16(fa1.v, fb0.v, acc[1][0]);
    acc[1][1] = WMMA_BF16(fa1.v, fb1.v, acc[1][1]);
    __syncthreads();   // all waves done reading `cur` before it is overwritten
  }

  #pragma unroll
  for (int ms = 0; ms < 2; ++ms) {
    #pragma unroll
    for (int r = 0; r < 8; ++r) {
      const int grow = rowBase + wm * 32 + ms * 16 + half * 8 + r;
      #pragma unroll
      for (int ns = 0; ns < 2; ++ns) {
        const int gcol = colBase + wn * 32 + ns * 16 + l16;
        float v = acc[ms][ns][r] + bias[gcol];
        if (residual) v += residual[(size_t)grow * N + gcol];
        if (do_relu)  v = fmaxf(v, 0.0f);
        if (Cf)  Cf[(size_t)grow * N + gcol] = v;
        if (Cb)  Cb[(size_t)grow * N + gcol] = (unsigned short)bf16b(v);
        if (CbT) CbT[((size_t)(grow >> 10) * N + gcol) * S_ + (grow & (S_ - 1))] =
                     (unsigned short)bf16b(v);
      }
    }
  }
}

// ---------------------------------------------------------------------------
// scores[bh,q,k] = mask(scale * Q_head . K_head^T). 128q x 64key block,
// waves 32x32, DK=64 = two K-steps -> 8 WMMAs/wave. Tiles via async-to-LDS.
// ---------------------------------------------------------------------------
__global__ void __launch_bounds__(256) attn_scores(
    const unsigned short* __restrict__ Qb, const unsigned short* __restrict__ Kb,
    const int* __restrict__ tokens, float* __restrict__ attnL, float scale)
{
  __shared__ __align__(16) unsigned short Qt[128 * 72];  // 144B row stride
  __shared__ __align__(16) unsigned short Kt[64 * 72];

  const int t = threadIdx.x;
  const int kBlk = blockIdx.x, qBlk = blockIdx.y, bh = blockIdx.z;
  const int b = bh >> 3, h = bh & (H_ - 1);
  const int qBase = qBlk * 128, keyBase = kBlk * 64;

  #pragma unroll
  for (int j = 0; j < 4; ++j) {                 // Q: 128 rows x 8 segs
    const int linear = t + j * 256;
    const int r = linear >> 3, seg = (linear & 7) << 3;
    async_copy_b128(Qb + (size_t)(b * S_ + qBase + r) * HD_ + h * DK_ + seg,
                    &Qt[r * 72 + seg]);
  }
  #pragma unroll
  for (int j = 0; j < 2; ++j) {                 // K: 64 rows x 8 segs
    const int linear = t + j * 256;
    const int r = linear >> 3, seg = (linear & 7) << 3;
    async_copy_b128(Kb + (size_t)(b * S_ + keyBase + r) * HD_ + h * DK_ + seg,
                    &Kt[r * 72 + seg]);
  }
  wait_async0();
  __syncthreads();

  const int lane = t & 31, wid = t >> 5;
  const int half = lane >> 4, l16 = lane & 15;
  const int wm = wid >> 1, wn = wid & 1;

  f32x8 acc[2][2] = {};
  #pragma unroll
  for (int kb = 0; kb < 2; ++kb) {
    const FragAB fa0 = frag_a(&Qt[(wm * 32 +      l16) * 72 + kb * 32], half);
    const FragAB fa1 = frag_a(&Qt[(wm * 32 + 16 + l16) * 72 + kb * 32], half);
    const FragAB fb0 = frag_b(&Kt[(wn * 32 +      l16) * 72 + kb * 32], half);
    const FragAB fb1 = frag_b(&Kt[(wn * 32 + 16 + l16) * 72 + kb * 32], half);
    acc[0][0] = WMMA_BF16(fa0.v, fb0.v, acc[0][0]);
    acc[0][1] = WMMA_BF16(fa0.v, fb1.v, acc[0][1]);
    acc[1][0] = WMMA_BF16(fa1.v, fb0.v, acc[1][0]);
    acc[1][1] = WMMA_BF16(fa1.v, fb1.v, acc[1][1]);
  }

  const size_t base = (size_t)bh * S_ * S_;
  #pragma unroll
  for (int ms = 0; ms < 2; ++ms) {
    #pragma unroll
    for (int r = 0; r < 8; ++r) {
      const int qrow = qBase + wm * 32 + ms * 16 + half * 8 + r;
      #pragma unroll
      for (int ns = 0; ns < 2; ++ns) {
        const int key = keyBase + wn * 32 + ns * 16 + l16;
        attnL[base + (size_t)qrow * S_ + key] =
            (tokens[b * S_ + key] == 0) ? -1e9f : acc[ms][ns][r] * scale;
      }
    }
  }
}

// ---------------------------------------------------------------------------
// Softmax over rows of S_: f32 in-place (attns output) + bf16 copy for attn.V
// ---------------------------------------------------------------------------
__global__ void __launch_bounds__(256) softmax_rows(
    float* __restrict__ attnL, unsigned short* __restrict__ attnB)
{
  __shared__ float red[256];
  const int t = threadIdx.x;
  float* p = attnL + (size_t)blockIdx.x * S_;
  unsigned short* pb = attnB + (size_t)blockIdx.x * S_;
  float v0 = p[t], v1 = p[t + 256], v2 = p[t + 512], v3 = p[t + 768];

  float m = fmaxf(fmaxf(v0, v1), fmaxf(v2, v3));
  red[t] = m; __syncthreads();
  for (int o = 128; o > 0; o >>= 1) {
    if (t < o) red[t] = fmaxf(red[t], red[t + o]);
    __syncthreads();
  }
  m = red[0]; __syncthreads();

  v0 = __expf(v0 - m); v1 = __expf(v1 - m);
  v2 = __expf(v2 - m); v3 = __expf(v3 - m);
  red[t] = v0 + v1 + v2 + v3; __syncthreads();
  for (int o = 128; o > 0; o >>= 1) {
    if (t < o) red[t] += red[t + o];
    __syncthreads();
  }
  const float inv = 1.0f / red[0];
  v0 *= inv; v1 *= inv; v2 *= inv; v3 *= inv;
  p[t] = v0; p[t + 256] = v1; p[t + 512] = v2; p[t + 768] = v3;
  pb[t]       = (unsigned short)bf16b(v0);
  pb[t + 256] = (unsigned short)bf16b(v1);
  pb[t + 512] = (unsigned short)bf16b(v2);
  pb[t + 768] = (unsigned short)bf16b(v3);
}

// ---------------------------------------------------------------------------
// ctx[b,q,hd] = sum_k attnB[bh,q,k] * Vt[b,hd,k]. 128q x 64d block, waves
// 32x32, K = S_ in 32-chunks; double-buffered async tiles like the GEMM.
// ---------------------------------------------------------------------------
__global__ void __launch_bounds__(256) attn_ctx(
    const unsigned short* __restrict__ attnB, const unsigned short* __restrict__ Vt,
    unsigned short* __restrict__ cxb)
{
  __shared__ __align__(16) unsigned short As[2][128 * 40];  // attn [q][key]
  __shared__ __align__(16) unsigned short Bs[2][64 * 40];   // Vt [d][key]

  const int t = threadIdx.x;
  const int qBlk = blockIdx.x, bh = blockIdx.y;
  const int b = bh >> 3, h = bh & (H_ - 1);
  const int qBase = qBlk * 128;
  const size_t abase = (size_t)bh * S_ * S_;

  const int lane = t & 31, wid = t >> 5;
  const int half = lane >> 4, l16 = lane & 15;
  const int wm = wid >> 1, wn = wid & 1;
  const int cr = t >> 2, cs = (t & 3) << 3;

  auto issue = [&](int kb, int buf) {
    const int kBase = kb << 5;
    async_copy_b128(attnB + abase + (size_t)(qBase + cr) * S_ + kBase + cs,
                    &As[buf][cr * 40 + cs]);
    async_copy_b128(attnB + abase + (size_t)(qBase + cr + 64) * S_ + kBase + cs,
                    &As[buf][(cr + 64) * 40 + cs]);
    async_copy_b128(Vt + (size_t)(b * HD_ + h * DK_ + cr) * S_ + kBase + cs,
                    &Bs[buf][cr * 40 + cs]);
  };

  f32x8 acc[2][2] = {};
  issue(0, 0);
  for (int kb = 0; kb < (S_ >> 5); ++kb) {
    const int cur = kb & 1;
    const bool more = (kb + 1 < (S_ >> 5));
    if (more) issue(kb + 1, cur ^ 1);
    if (more) wait_async3(); else wait_async0();
    __syncthreads();

    const FragAB fa0 = frag_a(&As[cur][(wm * 32 +      l16) * 40], half);
    const FragAB fa1 = frag_a(&As[cur][(wm * 32 + 16 + l16) * 40], half);
    const FragAB fb0 = frag_b(&Bs[cur][(wn * 32 +      l16) * 40], half);
    const FragAB fb1 = frag_b(&Bs[cur][(wn * 32 + 16 + l16) * 40], half);
    acc[0][0] = WMMA_BF16(fa0.v, fb0.v, acc[0][0]);
    acc[0][1] = WMMA_BF16(fa0.v, fb1.v, acc[0][1]);
    acc[1][0] = WMMA_BF16(fa1.v, fb0.v, acc[1][0]);
    acc[1][1] = WMMA_BF16(fa1.v, fb1.v, acc[1][1]);
    __syncthreads();
  }

  #pragma unroll
  for (int ms = 0; ms < 2; ++ms) {
    #pragma unroll
    for (int r = 0; r < 8; ++r) {
      const int qrow = qBase + wm * 32 + ms * 16 + half * 8 + r;
      #pragma unroll
      for (int ns = 0; ns < 2; ++ns) {
        const int d0 = wn * 32 + ns * 16 + l16;
        cxb[(size_t)(b * S_ + qrow) * HD_ + h * DK_ + d0] =
            (unsigned short)bf16b(acc[ms][ns][r]);
      }
    }
  }
}

// ---------------------------------------------------------------------------
// Row LayerNorm over D_=512: f32 in, bf16 out (feeds FFN1)
// ---------------------------------------------------------------------------
__global__ void __launch_bounds__(256) layernorm_rows(
    const float* __restrict__ in, const float* __restrict__ gamma,
    const float* __restrict__ beta, unsigned short* __restrict__ outb)
{
  __shared__ float red[256];
  const int t = threadIdx.x;
  const float* p = in + (size_t)blockIdx.x * D_;
  float a = p[t], b = p[t + 256];

  red[t] = a + b; __syncthreads();
  for (int o = 128; o > 0; o >>= 1) {
    if (t < o) red[t] += red[t + o];
    __syncthreads();
  }
  const float mean = red[0] * (1.0f / D_);
  __syncthreads();

  const float da = a - mean, db = b - mean;
  red[t] = da * da + db * db; __syncthreads();
  for (int o = 128; o > 0; o >>= 1) {
    if (t < o) red[t] += red[t + o];
    __syncthreads();
  }
  const float rstd = rsqrtf(red[0] * (1.0f / D_) + 1e-5f);

  unsigned short* q = outb + (size_t)blockIdx.x * D_;
  q[t]       = (unsigned short)bf16b(da * rstd * gamma[t]       + beta[t]);
  q[t + 256] = (unsigned short)bf16b(db * rstd * gamma[t + 256] + beta[t + 256]);
}

// ---------------------------------------------------------------------------
// Host-side orchestration (graph-capture safe)
// ---------------------------------------------------------------------------
extern "C" void kernel_launch(void* const* d_in, const int* in_sizes, int n_in,
                              void* d_out, int out_size, void* d_ws, size_t ws_size,
                              hipStream_t stream)
{
  (void)in_sizes; (void)n_in; (void)out_size; (void)ws_size;

  const int*   tokens = (const int*)  d_in[0];
  const float* emb    = (const float*)d_in[1];
  const float* Wq     = (const float*)d_in[2];
  const float* bq     = (const float*)d_in[3];
  const float* Wk     = (const float*)d_in[4];
  const float* bk     = (const float*)d_in[5];
  const float* Wv     = (const float*)d_in[6];
  const float* bv     = (const float*)d_in[7];
  const float* Wz     = (const float*)d_in[8];
  const float* bz     = (const float*)d_in[9];
  const float* gamma  = (const float*)d_in[10];
  const float* beta   = (const float*)d_in[11];
  const float* W1     = (const float*)d_in[12];
  const float* b1     = (const float*)d_in[13];
  const float* W2     = (const float*)d_in[14];
  const float* b2     = (const float*)d_in[15];

  float* out   = (float*)d_out;
  float* attns = out + (size_t)M_ * D_;            // [L,B,H,S,S]

  // Workspace layout (byte offsets, MB-aligned)
  char* w = (char*)d_ws;
  float*          x    = (float*)(w);                          // 4 MB f32 act
  float*          zpre = (float*)(w + (4ull  << 20));          // 4 MB pre-LN
  unsigned short* xb   = (unsigned short*)(w + (8ull  << 20)); // 2 MB bf16 act
  unsigned short* qb   = (unsigned short*)(w + (10ull << 20)); // 2 MB
  unsigned short* kbuf = (unsigned short*)(w + (12ull << 20)); // 2 MB
  unsigned short* vt   = (unsigned short*)(w + (14ull << 20)); // 2 MB  [b][hd][s]
  unsigned short* cxb  = (unsigned short*)(w + (16ull << 20)); // 2 MB
  unsigned short* zb   = (unsigned short*)(w + (18ull << 20)); // 2 MB
  unsigned short* h1b  = (unsigned short*)(w + (20ull << 20)); // 8 MB  [M][FF]
  unsigned short* attb = (unsigned short*)(w + (28ull << 20)); // 32 MB [bh][q][k]
  unsigned short* wqt  = (unsigned short*)(w + (60ull << 20)); // 0.5 MB [HD][D]
  unsigned short* wkt  = (unsigned short*)(w + (61ull << 20));
  unsigned short* wvt  = (unsigned short*)(w + (62ull << 20));
  unsigned short* wzt  = (unsigned short*)(w + (63ull << 20)); // [D][HD]
  unsigned short* w1t  = (unsigned short*)(w + (64ull << 20)); // 2 MB [FF][D]
  unsigned short* w2t  = (unsigned short*)(w + (66ull << 20)); // 2 MB [D][FF]

  embed_kernel<<<(M_ * D_) / 256, 256, 0, stream>>>(tokens, emb, x, xb);

  const dim3 blk(256);
  const dim3 gProj(D_ / 64,  M_ / 128);            // (8, 16)
  const dim3 gFF1 (FF_ / 64, M_ / 128);            // (32, 16)
  const dim3 gScr (S_ / 64, S_ / 128, B_ * H_);    // (16, 8, 16)
  const dim3 gCtx (S_ / 128, B_ * H_);             // (8, 16)

  for (int l = 0; l < L_; ++l) {
    const float* wq_l = Wq + (size_t)l * D_ * HD_;
    const float* wk_l = Wk + (size_t)l * D_ * HD_;
    const float* wv_l = Wv + (size_t)l * D_ * HD_;
    const float* wz_l = Wz + (size_t)l * HD_ * D_;
    const float* w1_l = W1 + (size_t)l * D_ * FF_;
    const float* w2_l = W2 + (size_t)l * FF_ * D_;
    const float* bq_l = bq + (size_t)l * HD_;
    const float* bk_l = bk + (size_t)l * HD_;
    const float* bv_l = bv + (size_t)l * HD_;
    const float* bz_l = bz + (size_t)l * D_;
    const float* g_l  = gamma + (size_t)l * D_;
    const float* be_l = beta  + (size_t)l * D_;
    const float* b1_l = b1 + (size_t)l * FF_;
    const float* b2_l = b2 + (size_t)l * D_;
    float* attnL = attns + (size_t)l * B_ * H_ * S_ * S_;

    // Weight convert+transpose to bf16 [N][K] (once per layer)
    convtrans_kernel<<<(D_ * HD_) / 256, blk, 0, stream>>>(wq_l, wqt, D_,  9);
    convtrans_kernel<<<(D_ * HD_) / 256, blk, 0, stream>>>(wk_l, wkt, D_,  9);
    convtrans_kernel<<<(D_ * HD_) / 256, blk, 0, stream>>>(wv_l, wvt, D_,  9);
    convtrans_kernel<<<(HD_ * D_) / 256, blk, 0, stream>>>(wz_l, wzt, HD_, 9);
    convtrans_kernel<<<(D_ * FF_) / 256, blk, 0, stream>>>(w1_l, w1t, D_, 11);
    convtrans_kernel<<<(FF_ * D_) / 256, blk, 0, stream>>>(w2_l, w2t, FF_, 9);

    // QKV projections (bf16 outputs; V written per-head transposed)
    gemm_bf16_wmma<<<gProj, blk, 0, stream>>>(xb, wqt, bq_l, nullptr,
                                              nullptr, qb,   nullptr, M_, HD_, D_, 0);
    gemm_bf16_wmma<<<gProj, blk, 0, stream>>>(xb, wkt, bk_l, nullptr,
                                              nullptr, kbuf, nullptr, M_, HD_, D_, 0);
    gemm_bf16_wmma<<<gProj, blk, 0, stream>>>(xb, wvt, bv_l, nullptr,
                                              nullptr, nullptr, vt,   M_, HD_, D_, 0);

    // Attention
    attn_scores<<<gScr, blk, 0, stream>>>(qb, kbuf, tokens, attnL, 0.125f);
    softmax_rows<<<B_ * H_ * S_, blk, 0, stream>>>(attnL, attb);
    attn_ctx<<<gCtx, blk, 0, stream>>>(attb, vt, cxb);

    // Output projection + residual (f32 x), then LayerNorm -> bf16 z
    gemm_bf16_wmma<<<gProj, blk, 0, stream>>>(cxb, wzt, bz_l, x,
                                              zpre, nullptr, nullptr, M_, D_, HD_, 0);
    layernorm_rows<<<M_, blk, 0, stream>>>(zpre, g_l, be_l, zb);

    // FFN: h1 = relu(z @ W1 + b1) (bf16); x_next = h1 @ W2 + b2 (f32 + bf16)
    gemm_bf16_wmma<<<gFF1, blk, 0, stream>>>(zb, w1t, b1_l, nullptr,
                                             nullptr, h1b, nullptr, M_, FF_, D_, 1);
    gemm_bf16_wmma<<<gProj, blk, 0, stream>>>(h1b, w2t, b2_l, nullptr,
                                              x, xb, nullptr, M_, D_, FF_, 0);
  }

  hipMemcpyAsync(out, x, sizeof(float) * M_ * D_, hipMemcpyDeviceToDevice, stream);
}
